// FeedbackModel_40346922779432
// MI455X (gfx1250) — compile-verified, matrix operand order checked by reading, not run on hardware
//
#include <hip/hip_runtime.h>
#include <hip/hip_bf16.h>

// ---------------------------------------------------------------------------
// LSTM warmup + autoregressive forecast for MI455X (gfx1250, wave32, WMMA).
//   UNITS=1024, FEATS=128, BATCH=256, WARMUP=256, FORECAST=128
// Strategy: bf16 WMMA (f32 accumulate) with weights pre-packed into exact
// v_wmma_f32_16x16x32_bf16 fragment layout; fused gate epilogue through LDS;
// double-buffered packed activations; decoder feeds predictions back packed.
// ---------------------------------------------------------------------------

#define UNITS    1024
#define FEATS    128
#define BATCH    256
#define WARMUP   256
#define FORECAST 128

#define KTOT (FEATS + UNITS)   // 1152 = x(128) | h(1024)
#define NKT  (KTOT / 32)       // 36 k-tiles of 32
#define NNT  (4 * UNITS / 16)  // 256 n-tiles of 16 (z width 4096)
#define NMT  (BATCH / 16)      // 16 m-tiles of 16

typedef __bf16 bf16;
typedef __attribute__((ext_vector_type(16))) __bf16 v16bf;
typedef __attribute__((ext_vector_type(8)))  float  v8f;

// K index held by element e (0..15) of a lane's fragment, hsel = lane>>4.
// (CDNA5 ISA 16-bit A 16x32 layout; B 32x16 uses the symmetric packing.)
__device__ __forceinline__ int kmap(int e, int hsel) {
    return ((e & 8) ? 16 : 0) + hsel * 8 + ((e >> 1) & 3) * 2 + (e & 1);
}

__device__ __forceinline__ bf16 tobf(float f) {
    union { float f; unsigned u; } v; v.f = f;
    unsigned r = v.u + 0x7FFFu + ((v.u >> 16) & 1u);   // round-to-nearest-even
    unsigned short h = (unsigned short)(r >> 16);
    return __builtin_bit_cast(__bf16, h);
}

__device__ __forceinline__ float sigm(float x) {
    return 1.0f / (1.0f + __expf(-x));
}

// ---------------------------------------------------------------------------
// One-time: pack [W_i;W_h] (K=1152, N=4096) and W_d (K=1024, N=128) into
// WMMA-B fragment order: [ntile][ktile][lane(32)][e(16)] bf16.
// ---------------------------------------------------------------------------
#define WP_TOTAL (NNT * NKT * 512)            // 4,718,592
#define WD_TOTAL ((UNITS / 32) * (FEATS / 16) * 512)  // 131,072

__global__ __launch_bounds__(256) void pack_w(const float* __restrict__ Wi,
                                              const float* __restrict__ Wh,
                                              const float* __restrict__ Wd,
                                              bf16* __restrict__ wp,
                                              bf16* __restrict__ wdp) {
    long tid = (long)blockIdx.x * 256 + threadIdx.x;
    if (tid < WP_TOTAL) {
        int e = tid & 15, lane = (tid >> 4) & 31;
        int j = (int)((tid >> 9) % NKT);
        int ng = (int)((tid >> 9) / NKT);
        int k = j * 32 + kmap(e, lane >> 4);
        int n = ng * 16 + (lane & 15);
        float v = (k < FEATS) ? Wi[(long)k * (4 * UNITS) + n]
                              : Wh[(long)(k - FEATS) * (4 * UNITS) + n];
        wp[tid] = tobf(v);
    } else {
        long t = tid - WP_TOTAL;
        if (t < WD_TOTAL) {
            int e = t & 15, lane = (t >> 4) & 31;
            int j = (int)((t >> 9) & 31);        // ktile 0..31
            int nt = (int)(t >> 14);             // ntile 0..7
            int k = j * 32 + kmap(e, lane >> 4);
            int n = nt * 16 + (lane & 15);
            wdp[t] = tobf(Wd[(long)k * FEATS + n]);
        }
    }
}

// ---------------------------------------------------------------------------
// Pack the x-region (ktiles 0..3) of the activation buffer from an f32 source
// (warmup inputs slice). 16 mtiles * 4 ktiles * 512 = 32768 elements.
// ---------------------------------------------------------------------------
__global__ __launch_bounds__(256) void pack_x(const float* __restrict__ xsrc,
                                              int xstride,
                                              bf16* __restrict__ Adst) {
    int tid = blockIdx.x * 256 + threadIdx.x;
    int e = tid & 15, lane = (tid >> 4) & 31;
    int j = (tid >> 9) & 3;
    int mt = tid >> 11;
    int m = mt * 16 + (lane & 15);
    int k = j * 32 + kmap(e, lane >> 4);
    Adst[(((long)mt * NKT + j) * 32 + lane) * 16 + e] =
        tobf(xsrc[(long)m * xstride + k]);
}

// ---------------------------------------------------------------------------
// Main recurrent step: z = A_packed * W_packed (+b), gates, c/h update.
// grid = (32 unit-groups, 4 m-groups), block = 256 (8 waves).
// Wave w: mtile = gy*4 + (w&3); gate-pair (w>>2); 4 accumulators.
// Fused epilogue via LDS exchange; writes c (f32) and h bf16-packed into Aout.
// ---------------------------------------------------------------------------
__global__ __launch_bounds__(256) void lstm_gemm(const bf16* __restrict__ Ain,
                                                 const bf16* __restrict__ Wp,
                                                 const float* __restrict__ b,
                                                 float* __restrict__ c,
                                                 bf16* __restrict__ Aout) {
    __shared__ float z[64 * 132];

    const int lane = threadIdx.x & 31;
    const int w    = threadIdx.x >> 5;
    const int mtl  = w & 3;
    const int gp   = w >> 2;                 // 0: gates i,f   1: gates g,o
    const int i    = blockIdx.y * 4 + mtl;   // global mtile
    const int u    = blockIdx.x;             // unit group (32 units)

    const int ng0 = (gp * 2 + 0) * 64 + u * 2;   // ntile: gate A, coltile 0
    const int ng2 = (gp * 2 + 1) * 64 + u * 2;   // ntile: gate B, coltile 0

    v8f acc0 = {0,0,0,0,0,0,0,0};
    v8f acc1 = {0,0,0,0,0,0,0,0};
    v8f acc2 = {0,0,0,0,0,0,0,0};
    v8f acc3 = {0,0,0,0,0,0,0,0};

    const bf16* Abase = Ain + ((long)i * NKT * 32 + lane) * 16;
    const bf16* B0    = Wp + (((long)(ng0 + 0) * NKT) * 32 + lane) * 16;
    const bf16* B1    = Wp + (((long)(ng0 + 1) * NKT) * 32 + lane) * 16;
    const bf16* B2    = Wp + (((long)(ng2 + 0) * NKT) * 32 + lane) * 16;
    const bf16* B3    = Wp + (((long)(ng2 + 1) * NKT) * 32 + lane) * 16;

    #pragma unroll 4
    for (int j = 0; j < NKT; ++j) {
        v16bf a  = *(const v16bf*)(Abase + (long)j * 512);
        v16bf b0 = *(const v16bf*)(B0    + (long)j * 512);
        v16bf b1 = *(const v16bf*)(B1    + (long)j * 512);
        v16bf b2 = *(const v16bf*)(B2    + (long)j * 512);
        v16bf b3 = *(const v16bf*)(B3    + (long)j * 512);
        acc0 = __builtin_amdgcn_wmma_f32_16x16x32_bf16(false, a, false, b0, (short)0, acc0, false, false);
        acc1 = __builtin_amdgcn_wmma_f32_16x16x32_bf16(false, a, false, b1, (short)0, acc1, false, false);
        acc2 = __builtin_amdgcn_wmma_f32_16x16x32_bf16(false, a, false, b2, (short)0, acc2, false, false);
        acc3 = __builtin_amdgcn_wmma_f32_16x16x32_bf16(false, a, false, b3, (short)0, acc3, false, false);
    }

    // Deposit z tiles into LDS: 64 rows x 128 gate-cols (stride 132).
    const int hsel = lane >> 4, nl = lane & 15;
    const int rowbase = mtl * 16 + 8 * hsel;
    const int cb0 = (gp * 2 + 0) * 32;       // gate A col base
    const int cb1 = (gp * 2 + 1) * 32;       // gate B col base
    #pragma unroll
    for (int r = 0; r < 8; ++r) {
        z[(rowbase + r) * 132 + cb0 +      nl] = acc0[r];
        z[(rowbase + r) * 132 + cb0 + 16 + nl] = acc1[r];
        z[(rowbase + r) * 132 + cb1 +      nl] = acc2[r];
        z[(rowbase + r) * 132 + cb1 + 16 + nl] = acc3[r];
    }
    __syncthreads();

    // Gate math: 64 rows x 32 units per block, 8 elements per thread.
    #pragma unroll
    for (int s = 0; s < 8; ++s) {
        int idx = threadIdx.x + s * 256;
        int rl  = idx >> 5;
        int ul  = idx & 31;
        int unit = u * 32 + ul;
        int m    = blockIdx.y * 64 + rl;

        float iv = z[rl * 132 +      ul] + b[unit];
        float fv = z[rl * 132 + 32 + ul] + b[UNITS + unit];
        float gv = z[rl * 132 + 64 + ul] + b[2 * UNITS + unit];
        float ov = z[rl * 132 + 96 + ul] + b[3 * UNITS + unit];

        long cidx = (long)m * UNITS + unit;
        float cn = sigm(fv) * c[cidx] + sigm(iv) * tanhf(gv);
        float hn = sigm(ov) * tanhf(cn);
        c[cidx] = cn;

        // Scatter h into packed-fragment layout of the next A buffer.
        int ka = FEATS + unit;
        int kt = ka >> 5, kk = ka & 31;
        int hs = (kk >> 3) & 1;
        int e  = ((kk >> 4) & 1) * 8 + (kk & 7);
        int ln = hs * 16 + (m & 15);
        int mt = m >> 4;
        Aout[(((long)mt * NKT + kt) * 32 + ln) * 16 + e] = tobf(hn);
    }
}

// ---------------------------------------------------------------------------
// Decoder: p = h @ W_d + b_d. Reads the packed h region (ktiles 4..35) of the
// current A buffer. Writes p f32 to out[:,t,:] AND scatters p bf16-packed
// into the x-region of the same buffer (feedback input for the next step).
// grid = 16 (mtiles), block = 256 (8 waves = 8 ntiles).
// ---------------------------------------------------------------------------
__global__ __launch_bounds__(256) void dense_k(const bf16* __restrict__ A,
                                               const bf16* __restrict__ Wdp,
                                               const float* __restrict__ bd,
                                               float* __restrict__ out,
                                               bf16* __restrict__ Axdst,
                                               int t) {
    const int lane = threadIdx.x & 31;
    const int w    = threadIdx.x >> 5;   // ntile 0..7
    const int i    = blockIdx.x;         // mtile 0..15

    v8f acc = {0,0,0,0,0,0,0,0};
    const bf16* Abase = A   + (((long)i * NKT + 4) * 32 + lane) * 16;
    const bf16* Bbase = Wdp + (((long)w * 32) * 32 + lane) * 16;

    #pragma unroll 4
    for (int j = 0; j < 32; ++j) {
        v16bf a  = *(const v16bf*)(Abase + (long)j * 512);
        v16bf bb = *(const v16bf*)(Bbase + (long)j * 512);
        acc = __builtin_amdgcn_wmma_f32_16x16x32_bf16(false, a, false, bb, (short)0, acc, false, false);
    }

    const int hsel = lane >> 4, nl = lane & 15;
    const int n = w * 16 + nl;
    const float bdn = bd[n];
    #pragma unroll
    for (int r = 0; r < 8; ++r) {
        int m = i * 16 + r + 8 * hsel;
        float p = acc[r] + bdn;
        out[((long)m * FORECAST + t) * FEATS + n] = p;

        int kt = n >> 5, kk = n & 31;
        int hs = (kk >> 3) & 1;
        int e  = ((kk >> 4) & 1) * 8 + (kk & 7);
        int ln = hs * 16 + (m & 15);
        int mt = m >> 4;
        Axdst[(((long)mt * NKT + kt) * 32 + ln) * 16 + e] = tobf(p);
    }
}

// ---------------------------------------------------------------------------
// Host orchestration (graph-captured): pack weights, zero state, run 256
// warmup steps then 128 forecast steps with double-buffered activations.
// ---------------------------------------------------------------------------
extern "C" void kernel_launch(void* const* d_in, const int* in_sizes, int n_in,
                              void* d_out, int out_size, void* d_ws, size_t ws_size,
                              hipStream_t stream) {
    (void)in_sizes; (void)n_in; (void)out_size; (void)ws_size;

    const float* inputs = (const float*)d_in[0];  // [256,256,128]
    const float* Wi     = (const float*)d_in[1];  // [128,4096]
    const float* Wh     = (const float*)d_in[2];  // [1024,4096]
    const float* b      = (const float*)d_in[3];  // [4096]
    const float* Wd     = (const float*)d_in[4];  // [1024,128]
    const float* bd     = (const float*)d_in[5];  // [128]
    float* out = (float*)d_out;

    char* ws = (char*)d_ws;
    const size_t WP_BYTES = (size_t)WP_TOTAL * 2;        // 9,437,184
    const size_t WD_BYTES = (size_t)WD_TOTAL * 2;        //   262,144
    const size_t A_BYTES  = (size_t)NMT * NKT * 512 * 2; //   589,824
    bf16*  Wp  = (bf16*)ws;
    bf16*  Wdp = (bf16*)(ws + WP_BYTES);
    bf16*  A0  = (bf16*)(ws + WP_BYTES + WD_BYTES);
    bf16*  A1  = (bf16*)(ws + WP_BYTES + WD_BYTES + A_BYTES);
    float* c   = (float*)(ws + WP_BYTES + WD_BYTES + 2 * A_BYTES);

    hipMemsetAsync(A0, 0, A_BYTES, stream);
    hipMemsetAsync(A1, 0, A_BYTES, stream);
    hipMemsetAsync(c,  0, (size_t)BATCH * UNITS * 4, stream);

    pack_w<<<(WP_TOTAL + WD_TOTAL) / 256, 256, 0, stream>>>(Wi, Wh, Wd, Wp, Wdp);

    bf16* Abuf[2] = {A0, A1};
    dim3 ggrid(32, 4);

    // Warmup: 256 sequential LSTM steps over the input sequence.
    for (int t = 0; t < WARMUP; ++t) {
        bf16* Acur  = Abuf[t & 1];
        bf16* Anext = Abuf[(t + 1) & 1];
        pack_x<<<128, 256, 0, stream>>>(inputs + (long)t * FEATS,
                                        WARMUP * FEATS, Acur);
        lstm_gemm<<<ggrid, 256, 0, stream>>>(Acur, Wp, b, c, Anext);
    }

    // Forecast: prediction feeds back as next input (packed in dense_k).
    int s = WARMUP;  // parity index of buffer holding the freshest h
    for (int t = 0; t < FORECAST; ++t) {
        bf16* Acur = Abuf[s & 1];
        dense_k<<<16, 256, 0, stream>>>(Acur, Wdp, bd, out, Acur, t);
        if (t < FORECAST - 1) {
            bf16* Anext = Abuf[(s + 1) & 1];
            lstm_gemm<<<ggrid, 256, 0, stream>>>(Acur, Wp, b, c, Anext);
            ++s;
        }
    }
}
